// PINN_87230785781834
// MI455X (gfx1250) — matrix operand order repaired and checked by reading, not compiled, not run on hardware
//
#include <hip/hip_runtime.h>
#include <math.h>

typedef __attribute__((ext_vector_type(16))) _Float16 v16h;
typedef __attribute__((ext_vector_type(8)))  _Float16 v8h;
typedef __attribute__((ext_vector_type(8)))  float    v8f;

// ---------------------------------------------------------------------------
// Prep: transpose W1..W3 (128x128, row-major [k][n]) into f16 WT[n][k] so B
// fragments (N striped across lanes, K contiguous) are 16-byte vector loads.
// ---------------------------------------------------------------------------
__global__ void prep_weights(const float* __restrict__ W1,
                             const float* __restrict__ W2,
                             const float* __restrict__ W3,
                             _Float16* __restrict__ wt) {
    int i = blockIdx.x * blockDim.x + threadIdx.x;   // i = k*128 + n
    if (i < 128 * 128) {
        int k = i >> 7, n = i & 127;
        int o = n * 128 + k;
        wt[0 * 16384 + o] = (_Float16)W1[i];
        wt[1 * 16384 + o] = (_Float16)W2[i];
        wt[2 * 16384 + o] = (_Float16)W3[i];
    }
}

// ---------------------------------------------------------------------------
// MLP + forward-mode tangents. One wave32 per 16-point tile.
// NCH: 1 = primal, 2 = primal + d/dy, 6 = primal,dx,dy,dt,dxx,dyy
// MODE: 0 = PDE residual, 1 = [u,v,p_y], 2 = [u,v,u], 3 = [u,v]
// ---------------------------------------------------------------------------
template <int NCH, int MODE>
__global__ __launch_bounds__(128)
void pinn_mlp(const float* __restrict__ xyt, int npts,
              const float* __restrict__ W0, const float* __restrict__ b0,
              const float* __restrict__ b1, const float* __restrict__ b2,
              const float* __restrict__ b3,
              const float* __restrict__ W4, const float* __restrict__ b4,
              const _Float16* __restrict__ WT,
              float* __restrict__ out, int out_base) {
    extern __shared__ _Float16 smem[];            // [nwaves][2][NCH][16][128]
    const int lane   = threadIdx.x & 31;
    const int wave   = threadIdx.x >> 5;
    const int nwaves = blockDim.x >> 5;
    const int lm     = lane & 15;                 // M/N sub-lane
    const int hi     = lane >> 4;                 // 0: lanes 0-15, 1: lanes 16-31

    _Float16* hbuf = smem + (size_t)wave * (2 * NCH * 2048);

    const int NT    = npts >> 4;                  // 16 points per tile
    const int wstep = gridDim.x * nwaves;

    for (int tile = blockIdx.x * nwaves + wave; tile < NT; tile += wstep) {
        // ---------------- layer 0: 3 -> 128 (scalar, tiny) ----------------
        _Float16* h0 = hbuf;                      // cur buffer = 0
        for (int idx = lane; idx < 16 * 128; idx += 32) {
            int m = idx >> 7, n = idx & 127;
            const float* q = xyt + (size_t)(tile * 16 + m) * 3;
            float zx = W0[n], zy = W0[128 + n], zt = W0[256 + n];
            float z  = q[0] * zx + q[1] * zy + q[2] * zt + b0[n];
            float s  = tanhf(z);
            float sp = 1.0f - s * s;
            h0[idx] = (_Float16)s;
            if (NCH == 2) {
                h0[2048 + idx] = (_Float16)(sp * zy);
            } else if (NCH == 6) {
                float spp = -2.0f * s * sp;
                h0[1 * 2048 + idx] = (_Float16)(sp * zx);
                h0[2 * 2048 + idx] = (_Float16)(sp * zy);
                h0[3 * 2048 + idx] = (_Float16)(sp * zt);
                h0[4 * 2048 + idx] = (_Float16)(spp * zx * zx);   // zxx = 0
                h0[5 * 2048 + idx] = (_Float16)(spp * zy * zy);   // zyy = 0
            }
        }
        int cur = 0;

        // ---------------- layers 1..3: 128 -> 128 via WMMA ----------------
        for (int layer = 0; layer < 3; ++layer) {
            const _Float16* wt   = WT + layer * 16384;
            const float*    bias = (layer == 0) ? b1 : (layer == 1) ? b2 : b3;
            const _Float16* hc   = hbuf + cur * (NCH * 2048);
            _Float16*       hn   = hbuf + (cur ^ 1) * (NCH * 2048);

            for (int t = 0; t < 8; ++t) {         // 16-wide output tiles
                float bn = bias[t * 16 + lm];
                v8f acc[NCH];
#pragma unroll
                for (int ch = 0; ch < NCH; ++ch)
#pragma unroll
                    for (int v = 0; v < 8; ++v) acc[ch][v] = (ch == 0) ? bn : 0.0f;

#pragma unroll
                for (int s = 0; s < 4; ++s) {     // K slices of 32
                    // B fragment: lane holds column n=16t+lm, 16 contiguous K
                    v16h bfrag;
                    const _Float16* bp = wt + (t * 16 + lm) * 128 + s * 32 + hi * 16;
                    ((v8h*)&bfrag)[0] = *(const v8h*)(bp);
                    ((v8h*)&bfrag)[1] = *(const v8h*)(bp + 8);
#pragma unroll
                    for (int ch = 0; ch < NCH; ++ch) {
                        // A fragment: lane holds row m=lm; lanes<16 K{0..15},
                        // lanes>=16 K{16..31} -> two ds_load_b128
                        v16h afrag;
                        const _Float16* ap = hc + ch * 2048 + lm * 128 + s * 32 + hi * 16;
                        ((v8h*)&afrag)[0] = *(const v8h*)(ap);
                        ((v8h*)&afrag)[1] = *(const v8h*)(ap + 8);
                        acc[ch] = __builtin_amdgcn_wmma_f32_16x16x32_f16(
                            false, afrag, false, bfrag, (short)0, acc[ch],
                            false, false);
                    }
                }

                // activation + tangent chain rule, store f16 to next buffer
#pragma unroll
                for (int v = 0; v < 8; ++v) {
                    int m = v + 8 * hi;
                    int o = m * 128 + t * 16 + lm;
                    float z  = acc[0][v];
                    float s  = tanhf(z);
                    float sp = 1.0f - s * s;
                    hn[o] = (_Float16)s;
                    if (NCH == 2) {
                        hn[2048 + o] = (_Float16)(sp * acc[1][v]);
                    } else if (NCH == 6) {
                        float spp = -2.0f * s * sp;
                        float zx = acc[1][v], zy = acc[2][v];
                        hn[1 * 2048 + o] = (_Float16)(sp * zx);
                        hn[2 * 2048 + o] = (_Float16)(sp * zy);
                        hn[3 * 2048 + o] = (_Float16)(sp * acc[3][v]);
                        hn[4 * 2048 + o] = (_Float16)(spp * zx * zx + sp * acc[4][v]);
                        hn[5 * 2048 + o] = (_Float16)(spp * zy * zy + sp * acc[5][v]);
                    }
                }
            }
            cur ^= 1;
        }

        // ---------------- layer 4: 128 -> 3 + epilogue ----------------
        const _Float16* hc = hbuf + cur * (NCH * 2048);
        int m    = lane >> 1;                     // 2 lanes per point
        int half = lane & 1;
        float part[NCH][3];
#pragma unroll
        for (int ch = 0; ch < NCH; ++ch)
            part[ch][0] = part[ch][1] = part[ch][2] = 0.0f;

        for (int k = 0; k < 64; ++k) {
            int kk = half * 64 + k;
            float w0 = W4[kk * 3 + 0], w1 = W4[kk * 3 + 1], w2 = W4[kk * 3 + 2];
#pragma unroll
            for (int ch = 0; ch < NCH; ++ch) {
                float hv = (float)hc[ch * 2048 + m * 128 + kk];
                part[ch][0] += hv * w0;
                part[ch][1] += hv * w1;
                part[ch][2] += hv * w2;
            }
        }
#pragma unroll
        for (int ch = 0; ch < NCH; ++ch)
#pragma unroll
            for (int j = 0; j < 3; ++j)
                part[ch][j] += __shfl_xor(part[ch][j], 1, 32);

        if (half == 0) {
            int   p = tile * 16 + m;
            float u = part[0][0] + b4[0];
            float v = part[0][1] + b4[1];
            if (MODE == 0) {
                // channels: 1=dx, 2=dy, 3=dt, 4=dxx, 5=dyy ; RHO=1, MU=0.01
                float ue = part[3][0] + u * part[1][0] + v * part[2][0]
                           + part[1][2] - 0.01f * (part[4][0] + part[5][0]);
                float ve = part[3][1] + u * part[1][1] + v * part[2][1]
                           + part[2][2] - 0.01f * (part[4][1] + part[5][1]);
                out[2 * p]     = ue;
                out[2 * p + 1] = ve;
            } else if (MODE == 1) {               // [u, v, p_y]
                out[out_base + 3 * p]     = u;
                out[out_base + 3 * p + 1] = v;
                out[out_base + 3 * p + 2] = part[1][2];
            } else if (MODE == 2) {               // [u, v, u]
                out[out_base + 3 * p]     = u;
                out[out_base + 3 * p + 1] = v;
                out[out_base + 3 * p + 2] = u;
            } else {                              // [u, v]
                out[out_base + 2 * p]     = u;
                out[out_base + 2 * p + 1] = v;
            }
        }
    }
}

// ---------------------------------------------------------------------------
extern "C" void kernel_launch(void* const* d_in, const int* in_sizes, int n_in,
                              void* d_out, int out_size, void* d_ws, size_t ws_size,
                              hipStream_t stream) {
    const float* xyt_eqn = (const float*)d_in[0];
    const float* xyt_in  = (const float*)d_in[1];
    /* d_in[2] = xyt_out, unused by the reference */
    const float* xyt_w1  = (const float*)d_in[3];
    const float* xyt_w2  = (const float*)d_in[4];
    const float* xyt_c   = (const float*)d_in[5];
    const float* xyt_i0  = (const float*)d_in[6];
    const float* W0 = (const float*)d_in[7];
    const float* b0 = (const float*)d_in[8];
    const float* W1 = (const float*)d_in[9];
    const float* b1 = (const float*)d_in[10];
    const float* W2 = (const float*)d_in[11];
    const float* b2 = (const float*)d_in[12];
    const float* W3 = (const float*)d_in[13];
    const float* b3 = (const float*)d_in[14];
    const float* W4 = (const float*)d_in[15];
    const float* b4 = (const float*)d_in[16];

    const int n_eqn = in_sizes[0] / 3;            // 131072
    const int n_b   = in_sizes[1] / 3;            // 8192

    _Float16* wt = (_Float16*)d_ws;               // 3*128*128 f16 = 98304 B
    float*    out = (float*)d_out;

    prep_weights<<<64, 256, 0, stream>>>(W1, W2, W3, wt);

    // flat output offsets (return order: PDE, uv_in, uv_w1, uv_w2, uv_circle, IC)
    const int off_in = n_eqn * 2;                 // 262144
    const int off_w1 = off_in + n_b * 2;          // 278528
    const int off_w2 = off_w1 + n_b * 3;          // 303104
    const int off_c  = off_w2 + n_b * 3;          // 327680
    const int off_ic = off_c + n_b * 3;           // 352256

    const size_t lds6 = (size_t)2 * 6 * 2048 * sizeof(_Float16) * 2;  // 2 waves
    const size_t lds2 = (size_t)2 * 2 * 2048 * sizeof(_Float16) * 4;  // 4 waves
    const size_t lds1 = (size_t)2 * 1 * 2048 * sizeof(_Float16) * 4;  // 4 waves

    pinn_mlp<6, 0><<<1024, 64, lds6, stream>>>(xyt_eqn, n_eqn, W0, b0, b1, b2, b3,
                                               W4, b4, wt, out, 0);
    pinn_mlp<2, 1><<<128, 128, lds2, stream>>>(xyt_w1, n_b, W0, b0, b1, b2, b3,
                                               W4, b4, wt, out, off_w1);
    pinn_mlp<2, 1><<<128, 128, lds2, stream>>>(xyt_w2, n_b, W0, b0, b1, b2, b3,
                                               W4, b4, wt, out, off_w2);
    pinn_mlp<1, 2><<<128, 128, lds1, stream>>>(xyt_c, n_b, W0, b0, b1, b2, b3,
                                               W4, b4, wt, out, off_c);
    pinn_mlp<1, 3><<<128, 128, lds1, stream>>>(xyt_in, n_b, W0, b0, b1, b2, b3,
                                               W4, b4, wt, out, off_in);
    pinn_mlp<1, 3><<<128, 128, lds1, stream>>>(xyt_i0, n_b, W0, b0, b1, b2, b3,
                                               W4, b4, wt, out, off_ic);
}
